// RGCN_90168543412868
// MI455X (gfx1250) — matrix operand order repaired and checked by reading, not compiled, not run on hardware
//
#include <hip/hip_runtime.h>
#include <hip/hip_bf16.h>

// ---------------------------------------------------------------------------
// RGCN on MI455X (gfx1250, wave32, WMMA).
// Strategy: transform-then-aggregate (mathematically identical to reference):
//   h = x@root + b;  for r: y = x@W[r];  for edges(type==r): h[dst] += ew*y[src]
// GEMMs: 64x128 block tile, bf16 LDS staging (W transposed), fragments loaded
// with ds_load_b128, 16x v_wmma_f32_16x16x32_bf16 per wave.
// Scatter: global_atomic_add_f32 into an L2-resident 25.6MB target.
// ---------------------------------------------------------------------------

typedef __attribute__((ext_vector_type(16))) __bf16 v16bf;
typedef __attribute__((ext_vector_type(8)))  float  v8f;

#define FDIM   128
#define BM     64            // rows per block
#define LDP    136           // padded LDS row stride in bf16 (272B, 16B-aligned)

union FragBF {
    v16bf v;
    uint4 q[2];
};

// ---- ordered-uint mapping for float atomic min/max ----
__device__ __forceinline__ unsigned f2key(float f) {
    unsigned b = __float_as_uint(f);
    return (b & 0x80000000u) ? ~b : (b | 0x80000000u);
}
__device__ __forceinline__ float key2f(unsigned k) {
    unsigned b = (k & 0x80000000u) ? (k & 0x7FFFFFFFu) : ~k;
    return __uint_as_float(b);
}

__global__ void rgcn_minmax_init(unsigned* mmx) {
    if (threadIdx.x == 0) { mmx[0] = 0xFFFFFFFFu; mmx[1] = 0u; }
}

__global__ __launch_bounds__(256)
void rgcn_minmax(const float* __restrict__ w, unsigned* mmx, int E) {
    __shared__ unsigned smn[256], smx[256];
    unsigned mn = 0xFFFFFFFFu, mx = 0u;
    for (int i = blockIdx.x * blockDim.x + threadIdx.x; i < E;
         i += gridDim.x * blockDim.x) {
        unsigned k = f2key(w[i]);
        mn = min(mn, k); mx = max(mx, k);
    }
    smn[threadIdx.x] = mn; smx[threadIdx.x] = mx;
    __syncthreads();
    for (int s = 128; s > 0; s >>= 1) {
        if (threadIdx.x < s) {
            smn[threadIdx.x] = min(smn[threadIdx.x], smn[threadIdx.x + s]);
            smx[threadIdx.x] = max(smx[threadIdx.x], smx[threadIdx.x + s]);
        }
        __syncthreads();
    }
    if (threadIdx.x == 0) {
        atomicMin(&mmx[0], smn[0]);
        atomicMax(&mmx[1], smx[0]);
    }
}

// ---------------------------------------------------------------------------
// GEMM: Y[row0..row0+63, 0..127] = X @ W(128x128 row-major KxO) (+bias)
// 8 waves in a 4x2 grid: wave = (rw, ch); each computes 16 rows x 64 cols
// as 4 accumulators. A-frag: 2x ds_load_b128 (reused over 4 n-tiles);
// B-frag: 2x ds_load_b128 from transposed W tile.
// ---------------------------------------------------------------------------
__global__ __launch_bounds__(256)
void rgcn_gemm(const float* __restrict__ X, const float* __restrict__ W,
               const float* __restrict__ bias, float* __restrict__ Y, int N) {
    __shared__ __bf16 sA[BM * LDP];      // x tile  [row][k], ~17 KB
    __shared__ __bf16 sBt[FDIM * LDP];   // weights [n][k] (transposed), ~34 KB

    const int tid  = threadIdx.x;
    const int row0 = blockIdx.x * BM;

    // Stage A tile: 64x128 elems, 32 per thread (coalesced over k)
    #pragma unroll
    for (int j = 0; j < 32; ++j) {
        int idx = tid + j * 256;             // idx = r*128 + c
        int r = idx >> 7, c = idx & 127;
        float v = (row0 + r < N) ? X[(size_t)(row0 + r) * FDIM + c] : 0.0f;
        sA[r * LDP + c] = (__bf16)v;
    }
    // Stage W transposed: read W[k][n] coalesced over n, write sBt[n][k]
    #pragma unroll
    for (int j = 0; j < 64; ++j) {
        int idx = tid + j * 256;             // idx = k*128 + n
        int k = idx >> 7, n = idx & 127;
        sBt[n * LDP + k] = (__bf16)W[idx];
    }
    __syncthreads();

    const int wid  = tid >> 5;
    const int lane = tid & 31;
    const int half = lane >> 4;          // 0: lanes 0-15, 1: lanes 16-31
    const int m    = lane & 15;
    const int rw   = wid >> 1;           // row tile 0..3
    const int ch   = wid & 1;            // col half 0..1

    v8f acc[4] = {};
    #pragma unroll
    for (int kb = 0; kb < 4; ++kb) {     // K blocks of 32
        // A (16x32): lane m = row; elems 0..7 -> K=base..base+7,
        //            elems 8..15 -> K=base+16..base+23 (ISA 7.12.2)
        FragBF a;
        const __bf16* ap = &sA[(rw * 16 + m) * LDP + kb * 32 + half * 8];
        a.q[0] = *(const uint4*)(ap);
        a.q[1] = *(const uint4*)(ap + 16);

        #pragma unroll
        for (int nt = 0; nt < 4; ++nt) {
            // B (32x16): lane m = col; elems 0..7 -> K = kb*32+half*8 .. +7,
            //            elems 8..15 -> K = kb*32+16+half*8 .. +7 (same as A)
            FragBF b;
            const __bf16* bp =
                &sBt[(ch * 64 + nt * 16 + m) * LDP + kb * 32 + half * 8];
            b.q[0] = *(const uint4*)(bp);
            b.q[1] = *(const uint4*)(bp + 16);
            acc[nt] = __builtin_amdgcn_wmma_f32_16x16x32_bf16(
                false, a.v, false, b.v, (short)0, acc[nt], false, false);
        }
    }

    // D layout: VGPR i -> row i (lanes 0-15) / i+8 (lanes 16-31), col = lane%16
    #pragma unroll
    for (int nt = 0; nt < 4; ++nt) {
        int col = ch * 64 + nt * 16 + m;
        float bv = bias ? bias[col] : 0.0f;
        #pragma unroll
        for (int i = 0; i < 8; ++i) {
            int row = row0 + rw * 16 + i + half * 8;
            if (row < N) Y[(size_t)row * FDIM + col] = acc[nt][i] + bv;
        }
    }
}

// ---------------------------------------------------------------------------
// Scatter: for edges with type==rel: H[dst] += ew_norm * Y[src]
// One wave32 per edge; each lane moves 4 floats (float4 gather + 4 f32 atomics).
// ---------------------------------------------------------------------------
__global__ __launch_bounds__(256)
void rgcn_scatter(const int* __restrict__ eidx, const int* __restrict__ etype,
                  const float* __restrict__ ew, const unsigned* __restrict__ mmx,
                  const float* __restrict__ Y, float* __restrict__ H,
                  int E, int rel) {
    int warp = (blockIdx.x * blockDim.x + threadIdx.x) >> 5;
    int lane = threadIdx.x & 31;
    if (warp >= E) return;
    if (etype[warp] != rel) return;

    int src = eidx[warp];
    int dst = eidx[E + warp];
    float mn = key2f(mmx[0]);
    float mx = key2f(mmx[1]);
    float wn = (ew[warp] - mn) / (mx - mn + 1e-8f);

    const float4* ys = (const float4*)(Y + (size_t)src * FDIM);
    float4 v = ys[lane];
    float* hd = H + (size_t)dst * FDIM + lane * 4;
    atomicAdd(hd + 0, wn * v.x);
    atomicAdd(hd + 1, wn * v.y);
    atomicAdd(hd + 2, wn * v.z);
    atomicAdd(hd + 3, wn * v.w);
}

__global__ __launch_bounds__(256)
void rgcn_relu(float* __restrict__ h, int n) {
    int i = blockIdx.x * blockDim.x + threadIdx.x;
    if (i < n) h[i] = fmaxf(h[i], 0.0f);
}

extern "C" void kernel_launch(void* const* d_in, const int* in_sizes, int n_in,
                              void* d_out, int out_size, void* d_ws, size_t ws_size,
                              hipStream_t stream) {
    const float* x     = (const float*)d_in[0];
    const int*   eidx  = (const int*)  d_in[1];
    const int*   etype = (const int*)  d_in[2];
    const float* ewt   = (const float*)d_in[3];
    const float* W1    = (const float*)d_in[4];
    const float* root1 = (const float*)d_in[5];
    const float* b1    = (const float*)d_in[6];
    const float* W2    = (const float*)d_in[7];
    const float* root2 = (const float*)d_in[8];
    const float* b2    = (const float*)d_in[9];
    float* out = (float*)d_out;

    const int N = in_sizes[0] / FDIM;
    const int E = in_sizes[2];
    const int R = in_sizes[4] / (FDIM * FDIM);

    // workspace: h [N*128] f32 | y [N*128] f32 | minmax keys (2 x u32)
    float*    h   = (float*)d_ws;
    float*    y   = h + (size_t)N * FDIM;
    unsigned* mmx = (unsigned*)(y + (size_t)N * FDIM);

    const int gemmBlocks    = (N + BM - 1) / BM;
    const int scatterBlocks = (E * 32 + 255) / 256;
    const int eltBlocks     = (N * FDIM + 255) / 256;

    // edge-weight min/max normalization constants (computed once)
    rgcn_minmax_init<<<1, 64, 0, stream>>>(mmx);
    rgcn_minmax<<<1024, 256, 0, stream>>>(ewt, mmx, E);

    // ---- layer 1: h = relu(x@root1 + b1 + sum_r scatter_r(x@W1[r])) ----
    rgcn_gemm<<<gemmBlocks, 256, 0, stream>>>(x, root1, b1, h, N);
    for (int r = 0; r < R; ++r) {
        rgcn_gemm<<<gemmBlocks, 256, 0, stream>>>(x, W1 + (size_t)r * FDIM * FDIM,
                                                  nullptr, y, N);
        rgcn_scatter<<<scatterBlocks, 256, 0, stream>>>(eidx, etype, ewt, mmx,
                                                        y, h, E, r);
    }
    rgcn_relu<<<eltBlocks, 256, 0, stream>>>(h, N * FDIM);

    // ---- layer 2: out = h@root2 + b2 + sum_r scatter_r(h@W2[r]) ----
    rgcn_gemm<<<gemmBlocks, 256, 0, stream>>>(h, root2, b2, out, N);
    for (int r = 0; r < R; ++r) {
        rgcn_gemm<<<gemmBlocks, 256, 0, stream>>>(h, W2 + (size_t)r * FDIM * FDIM,
                                                  nullptr, y, N);
        rgcn_scatter<<<scatterBlocks, 256, 0, stream>>>(eidx, etype, ewt, mmx,
                                                        y, out, E, r);
    }
}
